// GPT2_18872086299207
// MI455X (gfx1250) — matrix operand order repaired and checked by reading, not compiled, not run on hardware
//
#include <hip/hip_runtime.h>
#include <hip/hip_bf16.h>

// ---------------------------------------------------------------------------
// GPT-2 small forward on gfx1250 (CDNA5) using v_wmma_f32_16x16x32_f16.
// ---------------------------------------------------------------------------

#define LAYERS 12
#define HEADS  12
#define DIM    768
#define SEQ    1024
#define BATCH  2
#define VOCAB  50257
#define VPAD   50304      // VOCAB rounded up to multiple of 128
#define HD     64         // head dim
#define D3     (3*DIM)
#define D4     (4*DIM)

typedef _Float16 half8_t __attribute__((ext_vector_type(8)));
typedef _Float16 v16h    __attribute__((ext_vector_type(16)));
typedef float    v8f     __attribute__((ext_vector_type(8)));

#define FLAG_GELU   1
#define FLAG_CAUSAL 2

// Build a 16-half A/B fragment from two contiguous 8-half (16B) chunks.
__device__ __forceinline__ v16h ld_frag16(const _Float16* p0, const _Float16* p1) {
    union { v16h v; half8_t h[2]; } u;
    u.h[0] = *(const half8_t*)p0;
    u.h[1] = *(const half8_t*)p1;
    return u.v;
}

// ---------------------------------------------------------------------------
// Generic batched GEMM:  C[z] = A[z] (f16, [M,K] row-major)
//                              x Bt[z]^T (Bt is f16, [N,K] row-major = B^T)
//                              (+ bias[n]) (+ residual) (GELU / causal+scale)
// Block: 256 threads = 8 waves laid out 4(M) x 2(N).
// Wave tile: (MT*16) x (NT*16) via MT x NT v_wmma_f32_16x16x32_f16 tiles.
// Workgroup tile: (MT*64) x (NT*32). K stepped by 32.
// Requires: M % (MT*64) == 0, K % 32 == 0, Bt rows padded to N mult of NT*32.
// ---------------------------------------------------------------------------
template <int MT, int NT>
__global__ __launch_bounds__(256) void gemm_f16_kernel(
    const _Float16* __restrict__ A, const _Float16* __restrict__ Bt,
    const float* __restrict__ bias, const float* __restrict__ resid,
    float* __restrict__ C,
    int N, int K, int lda, int ldb, int ldc,
    long aBatch, long bBatch, long cBatch,
    int flags, float scale)
{
    const long z = blockIdx.z;
    A  += z * aBatch;
    Bt += z * bBatch;
    C  += z * cBatch;
    if (resid) resid += z * cBatch;

    const int lane = threadIdx.x & 31;
    const int wave = threadIdx.x >> 5;
    const int rowBase = blockIdx.y * (MT * 64) + (wave & 3) * (MT * 16);
    const int colBase = blockIdx.x * (NT * 32) + (wave >> 2) * (NT * 16);

    // Causal scores: tiles strictly above the diagonal are never read by the
    // softmax (it only touches j <= m) -> skip them. Wave-uniform branch.
    if ((flags & FLAG_CAUSAL) && (colBase > rowBase + MT * 16 - 1)) return;

    const int lrow = lane & 15;   // row (A) / col (B) within 16
    const int lgrp = lane >> 4;   // lane half-group

    const _Float16* aPtr[MT];
    const _Float16* bPtr[NT];
#pragma unroll
    for (int i = 0; i < MT; ++i) aPtr[i] = A  + (long)(rowBase + i * 16 + lrow) * lda;
#pragma unroll
    for (int j = 0; j < NT; ++j) bPtr[j] = Bt + (long)(colBase + j * 16 + lrow) * ldb;

    v8f acc[MT][NT];
#pragma unroll
    for (int i = 0; i < MT; ++i)
#pragma unroll
        for (int j = 0; j < NT; ++j) acc[i][j] = v8f{};

    for (int kt = 0; kt < K; kt += 32) {
        const int ka = kt + lgrp * 8;    // A: halves 0-7 @ ka, halves 8-15 @ ka+16
        const int kb = kt + lgrp * 8;    // B: halves 0-7 @ kb, halves 8-15 @ kb+16
        v16h af[MT], bf[NT];
#pragma unroll
        for (int i = 0; i < MT; ++i) af[i] = ld_frag16(aPtr[i] + ka, aPtr[i] + ka + 16);
#pragma unroll
        for (int j = 0; j < NT; ++j) bf[j] = ld_frag16(bPtr[j] + kb, bPtr[j] + kb + 16);
#pragma unroll
        for (int i = 0; i < MT; ++i)
#pragma unroll
            for (int j = 0; j < NT; ++j)
                acc[i][j] = __builtin_amdgcn_wmma_f32_16x16x32_f16(
                    false, af[i], false, bf[j], (short)0, acc[i][j], false, false);
    }

    // Epilogue. C/D layout: value r of a 16x16 tile sits at
    // m = mB + lgrp*8 + r, n = nB + lrow.
#pragma unroll
    for (int i = 0; i < MT; ++i) {
#pragma unroll
        for (int j = 0; j < NT; ++j) {
            const int n = colBase + j * 16 + lrow;
            if (n >= N) continue;
            const float bv = bias ? bias[n] : 0.0f;
            const int m0 = rowBase + i * 16 + lgrp * 8;
#pragma unroll
            for (int r = 0; r < 8; ++r) {
                const int m = m0 + r;
                float v = acc[i][j][r] + bv;
                if (flags & FLAG_CAUSAL) {
                    v *= scale;
                    if (n > m) v = -1e30f;
                }
                if (flags & FLAG_GELU) {
                    const float u = 0.7978845608028654f * (v + 0.044715f * v * v * v);
                    v = 0.5f * v * (1.0f + tanhf(u));
                }
                const long ci = (long)m * ldc + n;
                if (resid) v += resid[ci];
                C[ci] = v;
            }
        }
    }
}

// ---------------------------------------------------------------------------
// Embedding: h[b,t,:] = wte[x[b,t],:] + wpe[t,:]
// ---------------------------------------------------------------------------
__global__ __launch_bounds__(256) void embed_kernel(
    const int* __restrict__ x, const float* __restrict__ wte,
    const float* __restrict__ wpe, float* __restrict__ h)
{
    const long idx = (long)blockIdx.x * 256 + threadIdx.x;
    if (idx >= (long)BATCH * SEQ * DIM) return;
    const int d = (int)(idx % DIM);
    const int t = (int)((idx / DIM) % SEQ);
    const int b = (int)(idx / ((long)DIM * SEQ));
    const int tok = x[b * SEQ + t];
    h[idx] = wte[(long)tok * DIM + d] + wpe[(long)t * DIM + d];
}

// ---------------------------------------------------------------------------
// LayerNorm, fp32 stats, f16 output. One 256-thread block per row (D=768).
// ---------------------------------------------------------------------------
__global__ __launch_bounds__(256) void ln_kernel(
    const float* __restrict__ x, const float* __restrict__ w,
    const float* __restrict__ b, _Float16* __restrict__ out)
{
    __shared__ float s1[256], s2[256];
    const int row = blockIdx.x;
    const int tid = threadIdx.x;
    const float* xr = x + (long)row * DIM;
    float a = 0.f, a2 = 0.f;
    for (int i = tid; i < DIM; i += 256) { const float v = xr[i]; a += v; a2 += v * v; }
    s1[tid] = a; s2[tid] = a2;
    __syncthreads();
    for (int s = 128; s > 0; s >>= 1) {
        if (tid < s) { s1[tid] += s1[tid + s]; s2[tid] += s2[tid + s]; }
        __syncthreads();
    }
    const float mean = s1[0] * (1.0f / DIM);
    const float var  = s2[0] * (1.0f / DIM) - mean * mean;
    const float inv  = rsqrtf(var + 1e-5f);
    for (int i = tid; i < DIM; i += 256)
        out[(long)row * DIM + i] = (_Float16)((xr[i] - mean) * inv * w[i] + b[i]);
}

// ---------------------------------------------------------------------------
// Weight transpose + convert: W f32 [K,N] row-major -> Bt f16 [N,K].
// 32x32 LDS tile; K,N multiples of 32 for all uses.
// ---------------------------------------------------------------------------
__global__ __launch_bounds__(256) void transpose_w_kernel(
    const float* __restrict__ W, _Float16* __restrict__ Bt, int K, int N)
{
    __shared__ float tile[32][33];
    const int n0 = blockIdx.x * 32, k0 = blockIdx.y * 32;
    const int tx = threadIdx.x & 31, ty = threadIdx.x >> 5;
    for (int i = ty; i < 32; i += 8)
        tile[i][tx] = W[(long)(k0 + i) * N + (n0 + tx)];
    __syncthreads();
    for (int i = ty; i < 32; i += 8)
        Bt[(long)(n0 + i) * K + (k0 + tx)] = (_Float16)tile[tx][i];
}

// ---------------------------------------------------------------------------
// wte f32 [V,D] -> f16 [VPAD,D], zero rows V..VPAD (lm-head B matrix).
// ---------------------------------------------------------------------------
__global__ __launch_bounds__(256) void wte_f16_kernel(
    const float* __restrict__ wte, _Float16* __restrict__ out)
{
    const long idx = (long)blockIdx.x * 256 + threadIdx.x;
    if (idx >= (long)VPAD * DIM) return;
    const long n = idx / DIM;
    out[idx] = (n < VOCAB) ? (_Float16)wte[idx] : (_Float16)0.0f;
}

// ---------------------------------------------------------------------------
// Split QKV (fp32 [B,T,3D]) into per-head f16 buffers:
//   qf,kf : [B*H, T, HD]   (row-major, GEMM A / Bt form)
//   vt    : [B*H, HD, T]   (transposed so att x V uses the same GEMM)
// ---------------------------------------------------------------------------
__global__ __launch_bounds__(256) void split_qkv_kernel(
    const float* __restrict__ qkv, _Float16* __restrict__ qf,
    _Float16* __restrict__ kf, _Float16* __restrict__ vt)
{
    const long idx = (long)blockIdx.x * 256 + threadIdx.x;
    if (idx >= (long)BATCH * HEADS * SEQ * HD) return;
    const int d  = (int)(idx % HD);
    const int t  = (int)((idx / HD) % SEQ);
    const int hh = (int)((idx / ((long)HD * SEQ)) % HEADS);
    const int b  = (int)(idx / ((long)HD * SEQ * HEADS));
    const long src = ((long)b * SEQ + t) * D3 + hh * HD + d;
    const long zz  = (long)b * HEADS + hh;
    qf[(zz * SEQ + t) * HD + d]  = (_Float16)qkv[src];
    kf[(zz * SEQ + t) * HD + d]  = (_Float16)qkv[src + DIM];
    vt[(zz * HD + d) * SEQ + t]  = (_Float16)qkv[src + 2 * DIM];
}

// ---------------------------------------------------------------------------
// Causal softmax over S fp32 [B*H, T, T]; writes normalized probs f16
// (masked cols zeroed so the AV GEMM can run over full K=T).
// One 256-thread block per row; blockIdx.x = z*T + m.
// ---------------------------------------------------------------------------
__global__ __launch_bounds__(256) void softmax_kernel(
    const float* __restrict__ S, _Float16* __restrict__ P)
{
    __shared__ float red[256];
    const int rid = blockIdx.x;
    const int m   = rid & (SEQ - 1);
    const long base = (long)rid * SEQ;
    const int tid = threadIdx.x;

    float mx = -3.4e38f;
    for (int j = tid; j <= m; j += 256) mx = fmaxf(mx, S[base + j]);
    red[tid] = mx; __syncthreads();
    for (int s = 128; s > 0; s >>= 1) {
        if (tid < s) red[tid] = fmaxf(red[tid], red[tid + s]);
        __syncthreads();
    }
    mx = red[0]; __syncthreads();

    float sum = 0.f;
    for (int j = tid; j <= m; j += 256) sum += expf(S[base + j] - mx);
    red[tid] = sum; __syncthreads();
    for (int s = 128; s > 0; s >>= 1) {
        if (tid < s) red[tid] += red[tid + s];
        __syncthreads();
    }
    const float rinv = 1.0f / red[0];

    for (int j = tid; j < SEQ; j += 256) {
        const float v = (j <= m) ? expf(S[base + j] - mx) * rinv : 0.0f;
        P[base + j] = (_Float16)v;
    }
}

// ---------------------------------------------------------------------------
// Merge heads: yh fp32 [B*H, T, HD] -> f16 [B, T, D].
// ---------------------------------------------------------------------------
__global__ __launch_bounds__(256) void merge_heads_kernel(
    const float* __restrict__ yh, _Float16* __restrict__ out)
{
    const long idx = (long)blockIdx.x * 256 + threadIdx.x;
    if (idx >= (long)BATCH * SEQ * DIM) return;
    const int dd = (int)(idx % DIM);
    const int t  = (int)((idx / DIM) % SEQ);
    const int b  = (int)(idx / ((long)DIM * SEQ));
    const int hh = dd / HD, d = dd % HD;
    out[idx] = (_Float16)yh[(((long)b * HEADS + hh) * SEQ + t) * HD + d];
}

// ---------------------------------------------------------------------------
// f32 -> f16 elementwise convert.
// ---------------------------------------------------------------------------
__global__ __launch_bounds__(256) void f32_to_f16_kernel(
    const float* __restrict__ in, _Float16* __restrict__ out, long n)
{
    const long idx = (long)blockIdx.x * 256 + threadIdx.x;
    if (idx < n) out[idx] = (_Float16)in[idx];
}

// ---------------------------------------------------------------------------
// Host orchestration
// ---------------------------------------------------------------------------
extern "C" void kernel_launch(void* const* d_in, const int* in_sizes, int n_in,
                              void* d_out, int out_size, void* d_ws, size_t ws_size,
                              hipStream_t stream) {
    (void)in_sizes; (void)n_in; (void)out_size; (void)ws_size;

    const int*   x      = (const int*)  d_in[0];
    const float* wte    = (const float*)d_in[1];
    const float* wpe    = (const float*)d_in[2];
    const float* ln1_w  = (const float*)d_in[3];
    const float* ln1_b  = (const float*)d_in[4];
    const float* qkv_w  = (const float*)d_in[5];
    const float* qkv_b  = (const float*)d_in[6];
    const float* proj_w = (const float*)d_in[7];
    const float* proj_b = (const float*)d_in[8];
    const float* ln2_w  = (const float*)d_in[9];
    const float* ln2_b  = (const float*)d_in[10];
    const float* fc_w   = (const float*)d_in[11];
    const float* fc_b   = (const float*)d_in[12];
    const float* fcp_w  = (const float*)d_in[13];
    const float* fcp_b  = (const float*)d_in[14];
    const float* lnf_w  = (const float*)d_in[15];
    const float* lnf_b  = (const float*)d_in[16];
    float* logits = (float*)d_out;

    // ---- carve workspace ----
    char* p = (char*)d_ws;
    auto alloc = [&](size_t bytes) -> void* {
        void* r = (void*)p;
        p += (bytes + 255) & ~(size_t)255;
        return r;
    };
    const long MTOK = (long)BATCH * SEQ;          // 2048 rows
    _Float16* wteh  = (_Float16*)alloc((size_t)VPAD * DIM * 2);        // lm-head Bt
    _Float16* wbuf  = (_Float16*)alloc((size_t)D4 * DIM * 2);          // layer weight Bt (max 3072x768)
    float*    h     = (float*)   alloc((size_t)MTOK * DIM * 4);        // residual stream
    _Float16* lnh   = (_Float16*)alloc((size_t)MTOK * DIM * 2);        // LN output f16
    float*    qkv   = (float*)   alloc((size_t)MTOK * D3 * 4);
    _Float16* qf    = (_Float16*)alloc((size_t)BATCH * HEADS * SEQ * HD * 2);
    _Float16* kf    = (_Float16*)alloc((size_t)BATCH * HEADS * SEQ * HD * 2);
    _Float16* vt    = (_Float16*)alloc((size_t)BATCH * HEADS * HD * SEQ * 2);
    float*    S     = (float*)   alloc((size_t)BATCH * HEADS * SEQ * SEQ * 4);
    _Float16* P     = (_Float16*)alloc((size_t)BATCH * HEADS * SEQ * SEQ * 2);
    float*    yh    = (float*)   alloc((size_t)BATCH * HEADS * SEQ * HD * 4);
    _Float16* yhm   = (_Float16*)alloc((size_t)MTOK * DIM * 2);
    float*    mlp   = (float*)   alloc((size_t)MTOK * D4 * 4);
    _Float16* mlph  = (_Float16*)alloc((size_t)MTOK * D4 * 2);

    const int BH = BATCH * HEADS;
    const long eltBTD = (long)MTOK * DIM;

    // Workgroup tiles: <4,4> -> 256x128, <4,2> -> 256x64.
    auto grid44 = [](int Npad, int M, int batch) {
        return dim3((unsigned)(Npad / 128), (unsigned)(M / 256), (unsigned)batch);
    };
    auto grid42 = [](int Npad, int M, int batch) {
        return dim3((unsigned)(Npad / 64), (unsigned)(M / 256), (unsigned)batch);
    };

    // wte -> f16 padded (lm-head B matrix), once per call
    wte_f16_kernel<<<(unsigned)(((long)VPAD * DIM + 255) / 256), 256, 0, stream>>>(wte, wteh);

    // embeddings
    embed_kernel<<<(unsigned)((eltBTD + 255) / 256), 256, 0, stream>>>(x, wte, wpe, h);

    for (int l = 0; l < LAYERS; ++l) {
        // ---- attention ----
        ln_kernel<<<(unsigned)MTOK, 256, 0, stream>>>(h, ln1_w + l * DIM, ln1_b + l * DIM, lnh);

        transpose_w_kernel<<<dim3(D3 / 32, DIM / 32), 256, 0, stream>>>(
            qkv_w + (size_t)l * DIM * D3, wbuf, DIM, D3);
        gemm_f16_kernel<4, 4><<<grid44(D3, MTOK, 1), 256, 0, stream>>>(
            lnh, wbuf, qkv_b + l * D3, nullptr, qkv,
            D3, DIM, DIM, DIM, D3, 0, 0, 0, 0, 0.f);

        split_qkv_kernel<<<(unsigned)(((long)BH * SEQ * HD + 255) / 256), 256, 0, stream>>>(
            qkv, qf, kf, vt);

        // scores: S = (Q K^T) * scale, causal masked.  batch = B*H
        gemm_f16_kernel<4, 4><<<grid44(SEQ, SEQ, BH), 256, 0, stream>>>(
            qf, kf, nullptr, nullptr, S,
            SEQ, HD, HD, HD, SEQ,
            (long)SEQ * HD, (long)SEQ * HD, (long)SEQ * SEQ,
            FLAG_CAUSAL, 0.125f);

        softmax_kernel<<<(unsigned)(BH * SEQ), 256, 0, stream>>>(S, P);

        // yh = P @ V   (Bt = vt is [HD, T] per head)
        gemm_f16_kernel<4, 2><<<grid42(HD, SEQ, BH), 256, 0, stream>>>(
            P, vt, nullptr, nullptr, yh,
            HD, SEQ, SEQ, SEQ, HD,
            (long)SEQ * SEQ, (long)HD * SEQ, (long)SEQ * HD,
            0, 0.f);

        merge_heads_kernel<<<(unsigned)((eltBTD + 255) / 256), 256, 0, stream>>>(yh, yhm);

        // h += yhm @ proj_w + proj_b
        transpose_w_kernel<<<dim3(DIM / 32, DIM / 32), 256, 0, stream>>>(
            proj_w + (size_t)l * DIM * DIM, wbuf, DIM, DIM);
        gemm_f16_kernel<4, 4><<<grid44(DIM, MTOK, 1), 256, 0, stream>>>(
            yhm, wbuf, proj_b + l * DIM, h, h,
            DIM, DIM, DIM, DIM, DIM, 0, 0, 0, 0, 0.f);

        // ---- MLP ----
        ln_kernel<<<(unsigned)MTOK, 256, 0, stream>>>(h, ln2_w + l * DIM, ln2_b + l * DIM, lnh);

        transpose_w_kernel<<<dim3(D4 / 32, DIM / 32), 256, 0, stream>>>(
            fc_w + (size_t)l * DIM * D4, wbuf, DIM, D4);
        gemm_f16_kernel<4, 4><<<grid44(D4, MTOK, 1), 256, 0, stream>>>(
            lnh, wbuf, fc_b + l * D4, nullptr, mlp,
            D4, DIM, DIM, DIM, D4, 0, 0, 0, FLAG_GELU, 0.f);

        f32_to_f16_kernel<<<(unsigned)(((long)MTOK * D4 + 255) / 256), 256, 0, stream>>>(
            mlp, mlph, (long)MTOK * D4);

        transpose_w_kernel<<<dim3(DIM / 32, D4 / 32), 256, 0, stream>>>(
            fcp_w + (size_t)l * D4 * DIM, wbuf, D4, DIM);
        gemm_f16_kernel<4, 4><<<grid44(DIM, MTOK, 1), 256, 0, stream>>>(
            mlph, wbuf, fcp_b + l * DIM, h, h,
            DIM, D4, D4, D4, DIM, 0, 0, 0, 0, 0.f);
    }

    // final LN + tied lm-head: logits = ln_f(h) @ wte^T
    ln_kernel<<<(unsigned)MTOK, 256, 0, stream>>>(h, lnf_w, lnf_b, lnh);
    gemm_f16_kernel<4, 4><<<grid44(VPAD, MTOK, 1), 256, 0, stream>>>(
        lnh, wteh, nullptr, nullptr, logits,
        VOCAB, DIM, DIM, DIM, VOCAB, 0, 0, 0, 0, 0.f);
}
